// Attention_81398220193933
// MI455X (gfx1250) — matrix operand order repaired and checked by reading, not compiled, not run on hardware
//
#include <hip/hip_runtime.h>

#define UNITS 1024
#define HEADS 16
#define DH    64
#define SEQ   1024
#define BATCH 8
#define KT    64          // keys per tile (4 x 16-key sub-tiles)
#define QT    16          // queries per wave
#define WAVES 4
#define QBLK  (QT * WAVES)
#define MASK_VALUE (-1.0e12f)
#define LOG2E 1.4426950408889634f

typedef __attribute__((ext_vector_type(16))) _Float16 v16h;
typedef __attribute__((ext_vector_type(8)))  float    v8f;
typedef __attribute__((ext_vector_type(2)))  _Float16 v2h;

union F16F {
    v16h  v;
    v2h   d[8];
    uint4 q[2];
};

union H4U2 {
    _Float16 h[4];
    v2h      d[2];
    uint2    u;
};

// f32 pair -> packed f16 (v_cvt_pk_rtz_f16_f32) with scalar fallback
__device__ __forceinline__ v2h cvt2h(float a, float b) {
#if __has_builtin(__builtin_amdgcn_cvt_pkrtz)
    return __builtin_bit_cast(v2h, __builtin_amdgcn_cvt_pkrtz(a, b));
#else
    v2h r; r[0] = (_Float16)a; r[1] = (_Float16)b; return r;
#endif
}

// hardware exp2 (v_exp_f32)
__device__ __forceinline__ float fast_exp2(float x) {
#if __has_builtin(__builtin_amdgcn_exp2f)
    return __builtin_amdgcn_exp2f(x);
#else
    return exp2f(x);
#endif
}

// PV-contraction key permutation: sigma(key) = (key % 16) * 4 + key / 16.
__device__ __forceinline__ int sigma_key(int key) {
    return ((key & 15) << 2) | (key >> 4);
}

#if __has_builtin(__builtin_amdgcn_permlane16)
__device__ __forceinline__ float permx16(float v, unsigned s0, unsigned s1) {
    return __uint_as_float(__builtin_amdgcn_permlane16(
        __float_as_uint(v), __float_as_uint(v), s0, s1, false, false));
}
#endif

__device__ __forceinline__ float red_max16(float v) {
#if __has_builtin(__builtin_amdgcn_permlane16)
    v = fmaxf(v, permx16(v, 0x67452301u, 0xEFCDAB89u));   // xor 1
    v = fmaxf(v, permx16(v, 0x54761032u, 0xDCFE98BAu));   // xor 2
    v = fmaxf(v, permx16(v, 0x32107654u, 0xBA98FEDCu));   // xor 4
    v = fmaxf(v, permx16(v, 0xFEDCBA98u, 0x76543210u));   // xor 8
#else
    #pragma unroll
    for (int off = 1; off < 16; off <<= 1)
        v = fmaxf(v, __shfl_xor(v, off, 32));
#endif
    return v;
}

__device__ __forceinline__ float red_sum16(float v) {
#if __has_builtin(__builtin_amdgcn_permlane16)
    v += permx16(v, 0x67452301u, 0xEFCDAB89u);
    v += permx16(v, 0x54761032u, 0xDCFE98BAu);
    v += permx16(v, 0x32107654u, 0xBA98FEDCu);
    v += permx16(v, 0xFEDCBA98u, 0x76543210u);
#else
    #pragma unroll
    for (int off = 1; off < 16; off <<= 1)
        v += __shfl_xor(v, off, 32);
#endif
    return v;
}

__global__ __launch_bounds__(128, 2)
void fa_fwd_kernel(const float* __restrict__ memory,
                   const float* __restrict__ query,
                   const float* __restrict__ bias,
                   const int*   __restrict__ seq_len,
                   float* __restrict__ out)
{
    __shared__ _Float16 Kt[KT * DH];          // [key][dh]            8KB
    __shared__ _Float16 Vt[DH * KT];          // [dh][sigma(key)]     8KB
    __shared__ _Float16 Pb[WAVES * QT * KT];  // [row][sigma(key)]    8KB

    const int tid  = threadIdx.x;
    const int lane = tid & 31;
    const int wave = tid >> 5;
    const int lh   = lane >> 4;
    const int ln   = lane & 15;

    const int qb = blockIdx.x * QBLK + wave * QT;
    const int h  = blockIdx.y;
    const int b  = blockIdx.z;

    const int seq = seq_len[b];

    const float* kg = memory + (size_t)b * SEQ * (2 * UNITS) + h * DH;  // K half
    const float* vg = kg + UNITS;                                       // V half
    const float* qg = query + ((size_t)b * SEQ + qb) * UNITS + h * DH;

    // ---- Q fragments: fp32 -> f16, scaled by dh^-0.5 * log2(e)  ----
    // (base-2 logit domain: exp(x) becomes a single v_exp_f32)
    const float qscale = 0.125f * LOG2E;
    v16h qa[2];
    {
        const float* qrow = qg + (size_t)ln * UNITS;
        #pragma unroll
        for (int dc = 0; dc < 2; ++dc) {
            F16F f;
            const int d0 = dc * 32 + lh * 8;
            #pragma unroll
            for (int i = 0; i < 4; ++i) {
                f.d[i]     = cvt2h(qrow[d0 + 2*i]      * qscale, qrow[d0 + 2*i + 1]      * qscale);
                f.d[4 + i] = cvt2h(qrow[d0 + 16 + 2*i] * qscale, qrow[d0 + 16 + 2*i + 1] * qscale);
            }
            qa[dc] = f.v;
        }
    }

    float m[8], lpart[8];
    #pragma unroll
    for (int r = 0; r < 8; ++r) { m[r] = -3.0e38f; lpart[r] = 0.0f; }
    v8f acc[4] = {};

    #pragma unroll 1
    for (int kt = 0; kt < SEQ / KT; ++kt) {
        const int k0 = kt * KT;
        __syncthreads();   // prev-iteration tile reads complete

        // ---- cooperative stage of 64-key K / V tile (fp32 -> f16) ----
        #pragma unroll
        for (int j = 0; j < 2; ++j) {
            const int task = tid + j * 128;
            const int row  = task >> 2;          // key row 0..63
            const int cg   = (task & 3) * 16;    // 16-float col group
            const float4* kr = (const float4*)(kg + (size_t)(k0 + row) * (2 * UNITS) + cg);
            const float4* vr = (const float4*)(vg + (size_t)(k0 + row) * (2 * UNITS) + cg);
            float4 ka = kr[0], kb = kr[1], kc = kr[2], kd = kr[3];
            float4 va = vr[0], vb = vr[1], vc = vr[2], vd = vr[3];
            F16F kf;
            kf.d[0] = cvt2h(ka.x, ka.y); kf.d[1] = cvt2h(ka.z, ka.w);
            kf.d[2] = cvt2h(kb.x, kb.y); kf.d[3] = cvt2h(kb.z, kb.w);
            kf.d[4] = cvt2h(kc.x, kc.y); kf.d[5] = cvt2h(kc.z, kc.w);
            kf.d[6] = cvt2h(kd.x, kd.y); kf.d[7] = cvt2h(kd.z, kd.w);
            *(uint4*)&Kt[row * DH + cg]     = kf.q[0];
            *(uint4*)&Kt[row * DH + cg + 8] = kf.q[1];
            const int vrow = sigma_key(row);
            float vv[16] = { va.x, va.y, va.z, va.w, vb.x, vb.y, vb.z, vb.w,
                             vc.x, vc.y, vc.z, vc.w, vd.x, vd.y, vd.z, vd.w };
            #pragma unroll
            for (int i = 0; i < 16; ++i)
                Vt[(cg + i) * KT + vrow] = (_Float16)vv[i];  // transposed + permuted
            if (kt + 1 < SEQ / KT) {
                __builtin_prefetch((const float*)kr + KT * (2 * UNITS), 0, 0);
                __builtin_prefetch((const float*)vr + KT * (2 * UNITS), 0, 0);
            }
        }
        __syncthreads();

        // ---- bias + length mask, base-2 domain, folded into WMMA C ----
        float bm[4];
        #pragma unroll
        for (int nh = 0; nh < 4; ++nh) {
            const int key = k0 + nh * 16 + ln;
            bm[nh] = bias[key] * LOG2E + (key < seq ? 0.0f : MASK_VALUE * LOG2E);
        }

        // ---- S = Q K^T + bias + mask (C initialized to bm, lane-uniform) ----
        v8f s[4];
        #pragma unroll
        for (int nh = 0; nh < 4; ++nh) {
            v8f c;
            #pragma unroll
            for (int r = 0; r < 8; ++r) c[r] = bm[nh];
            #pragma unroll
            for (int dc = 0; dc < 2; ++dc) {
                F16F bf;
                const _Float16* p = &Kt[(nh * 16 + ln) * DH + dc * 32 + lh * 8];
                bf.q[0] = *(const uint4*)p;
                bf.q[1] = *(const uint4*)(p + 16);
                c = __builtin_amdgcn_wmma_f32_16x16x32_f16(
                        false, qa[dc], false, bf.v, (short)0, c, false, false);
            }
            s[nh] = c;
        }

        // ---- online softmax (base-2): row-max butterfly via v_permlane16 ----
        #pragma unroll
        for (int r = 0; r < 8; ++r) {
            const float mx = red_max16(
                fmaxf(fmaxf(s[0][r], s[1][r]), fmaxf(s[2][r], s[3][r])));
            const float mnew = fmaxf(m[r], mx);
            const float corr = fast_exp2(m[r] - mnew);     // row-uniform
            m[r] = mnew;
            const float p0 = fast_exp2(s[0][r] - mnew);
            const float p1 = fast_exp2(s[1][r] - mnew);
            const float p2 = fast_exp2(s[2][r] - mnew);
            const float p3 = fast_exp2(s[3][r] - mnew);
            lpart[r] = lpart[r] * corr + ((p0 + p1) + (p2 + p3));
            #pragma unroll
            for (int t = 0; t < 4; ++t) acc[t][r] *= corr;
            H4U2 pk;
            pk.d[0] = cvt2h(p0, p1);                       // keys ln, 16+ln
            pk.d[1] = cvt2h(p2, p3);                       // keys 32+ln, 48+ln
            const int prow = (wave * QT + r + 8 * lh) * KT;
            *(uint2*)&Pb[prow + 4 * ln] = pk.u;            // one ds_store_b64
        }
        // P is wave-private; same-wave LDS ops execute in order (ISA 7.3),
        // so only a compiler reordering fence is needed here (no s_barrier).
        asm volatile("" ::: "memory");

        // ---- O += P V : K-dim = 64 (permuted) keys as 2 x K32 chunks ----
        #pragma unroll
        for (int kc = 0; kc < 2; ++kc) {
            F16F pf;
            {
                const _Float16* p = &Pb[(wave * QT + ln) * KT + kc * 32 + lh * 8];
                pf.q[0] = *(const uint4*)p;
                pf.q[1] = *(const uint4*)(p + 16);
            }
            #pragma unroll
            for (int t = 0; t < 4; ++t) {
                F16F vf;
                const _Float16* p = &Vt[(t * 16 + ln) * KT + kc * 32 + lh * 8];
                vf.q[0] = *(const uint4*)p;
                vf.q[1] = *(const uint4*)(p + 16);
                acc[t] = __builtin_amdgcn_wmma_f32_16x16x32_f16(
                             false, pf.v, false, vf.v, (short)0, acc[t], false, false);
            }
        }
    }

    // ---- deferred row-sum reduction, finalize, write fp32 output ----
    #pragma unroll
    for (int r = 0; r < 8; ++r) {
        const float inv = 1.0f / red_sum16(lpart[r]);
        const int q = qb + r + 8 * lh;
        float* orow = out + ((size_t)b * SEQ + q) * UNITS + h * DH;
        #pragma unroll
        for (int t = 0; t < 4; ++t)
            orow[t * 16 + ln] = acc[t][r] * inv;
    }
}

extern "C" void kernel_launch(void* const* d_in, const int* in_sizes, int n_in,
                              void* d_out, int out_size, void* d_ws, size_t ws_size,
                              hipStream_t stream) {
    const float* memory = (const float*)d_in[0];
    const float* query  = (const float*)d_in[1];
    const float* bias   = (const float*)d_in[2];
    const int*   seqlen = (const int*)d_in[3];
    float* out = (float*)d_out;

    dim3 grid(SEQ / QBLK, HEADS, BATCH);   // 16 x 16 x 8 = 2048 workgroups
    fa_fwd_kernel<<<grid, 128, 0, stream>>>(memory, query, bias, seqlen, out);
}
